// RGCNLayer_88029649699360
// MI455X (gfx1250) — matrix operand order repaired and checked by reading, not compiled, not run on hardware
//
#include <hip/hip_runtime.h>
#include <hip/hip_bf16.h>

typedef __attribute__((ext_vector_type(16))) __bf16 v16bf;
typedef __attribute__((ext_vector_type(8)))  float  v8f;

#define D 128
#define AROWS 64
#define LSTRIDE 136   // ushort stride: 272B rows -> 16B aligned, bank-conflict-free

#if defined(__has_builtin)
#  if __has_builtin(__builtin_amdgcn_global_load_async_to_lds_b128)
#    define HAVE_ASYNC_LDS 1    // feature-detect: gfx1250 async-to-LDS path exists
#  endif
#endif

typedef __attribute__((address_space(3))) unsigned short* lds_us_ptr;

__device__ __forceinline__ unsigned short f2bf(float f) {
    unsigned int u = __float_as_uint(f);
    u += 0x7FFFu + ((u >> 16) & 1u);   // round-to-nearest-even
    return (unsigned short)(u >> 16);
}

union BF16x16 { v16bf v; uint4 q[2]; };

// ---- W (fp32 [r][k][n]) -> Wt (bf16 [r][n][k]), run once per launch ----
__global__ __launch_bounds__(256) void convw_kernel(const float* __restrict__ W,
                                                    unsigned short* __restrict__ Wt,
                                                    int total) {
    int idx = blockIdx.x * blockDim.x + threadIdx.x;
    if (idx >= total) return;
    int n = idx & 127;
    int k = (idx >> 7) & 127;
    int r = idx >> 14;
    Wt[((size_t)r * D + n) * D + k] = f2bf(W[idx]);
}

// ---- degree counts: out_deg over src, in_deg over dst ----
__global__ __launch_bounds__(256) void deg_kernel(const int* __restrict__ src,
                                                  const int* __restrict__ dst,
                                                  float* outdeg, float* indeg, int E) {
    int e = blockIdx.x * blockDim.x + threadIdx.x;
    if (e < E) {
        atomicAdd(&outdeg[src[e]], 1.0f);
        atomicAdd(&indeg[dst[e]], 1.0f);
    }
}

// ---- agg[dst] += features[src] * out_deg^-1/2 ; one wave per edge ----
__global__ __launch_bounds__(256) void scatter_kernel(const float* __restrict__ feat,
                                                      const int* __restrict__ src,
                                                      const int* __restrict__ dst,
                                                      const float* __restrict__ outdeg,
                                                      float* agg, int E) {
    int tid = blockIdx.x * blockDim.x + threadIdx.x;
    int e = tid >> 5;
    if (e >= E) return;
    int c = (tid & 31) << 2;                     // 32 lanes x float4 = one 128-f row
    int s = src[e], d = dst[e];
    float sc = rsqrtf(fmaxf(outdeg[s], 1.0f));
    const float4 f = *(const float4*)(feat + (size_t)s * D + c);
    float* p = agg + (size_t)d * D + c;
    atomicAdd(p + 0, f.x * sc);
    atomicAdd(p + 1, f.y * sc);
    atomicAdd(p + 2, f.z * sc);
    atomicAdd(p + 3, f.w * sc);
}

// ---- out += (agg @ W[r]) * in_deg^-1/2 + b[r], bf16 WMMA, f32 accum ----
__global__ __launch_bounds__(256) void gemm_kernel(const float* __restrict__ agg,
                                                   const unsigned short* __restrict__ Wt, // bf16 [n][k]
                                                   const float* __restrict__ br,
                                                   const float* __restrict__ indeg,
                                                   float* __restrict__ out, int N) {
    __shared__ __align__(16) unsigned short Alds[AROWS * LSTRIDE]; // [row][k] bf16
    __shared__ __align__(16) unsigned short Blds[D * LSTRIDE];     // [n][k]  bf16

    const int t = threadIdx.x;
    const int blockRow = blockIdx.x * AROWS;

    // Stage B: bf16 [n][k] -> LDS. 2048 16B chunks total -> 8 per thread.
#if defined(HAVE_ASYNC_LDS)
    #pragma unroll
    for (int i = 0; i < (D * D / 8) / 256; ++i) {      // 8 chunks per thread
        int id2 = t + 256 * i;                          // 0..2047
        int row = id2 >> 4, seg = id2 & 15;
        unsigned ldsoff =
            (unsigned)(unsigned long long)(lds_us_ptr)(&Blds[row * LSTRIDE + seg * 8]);
        unsigned long long gaddr =
            (unsigned long long)(const void*)(Wt + (size_t)row * D + seg * 8);
        asm volatile("global_load_async_to_lds_b128 %0, %1, off"
                     :: "v"(ldsoff), "v"(gaddr) : "memory");
    }
#else
    #pragma unroll
    for (int i = 0; i < (D * D / 8) / 256; ++i) {
        int id2 = t + 256 * i;
        int row = id2 >> 4, seg = id2 & 15;
        *(uint4*)(&Blds[row * LSTRIDE + seg * 8]) =
            *(const uint4*)(Wt + (size_t)row * D + seg * 8);
    }
#endif

    // Stage A tile fp32 -> bf16 (coalesced global reads)
    for (int i = 0; i < (AROWS * D) / 256; ++i) {
        int idx = t + 256 * i;
        int row = idx >> 7, k = idx & 127;
        int gr = blockRow + row;
        float v = (gr < N) ? agg[(size_t)gr * D + k] : 0.0f;
        Alds[row * LSTRIDE + k] = f2bf(v);
    }

#if defined(HAVE_ASYNC_LDS)
    asm volatile("s_wait_asynccnt 0" ::: "memory");
#endif
    __syncthreads();

    const int lane    = t & 31;
    const int w       = t >> 5;
    const int rowTile = w & 3;          // 4 x 16-row tiles
    const int colBase = (w >> 2) * 64;  // 2 column halves
    const int hi      = lane >> 4;      // lane half

    // A fragments: lane L -> row (L&15); K pairs per ISA 16-bit A layout
    const int m  = (rowTile << 4) + (lane & 15);
    const int kA = hi ? 8 : 0;
    v16bf afrag[4];
    #pragma unroll
    for (int kb = 0; kb < 4; ++kb) {
        BF16x16 u;
        const unsigned short* p = Alds + m * LSTRIDE + kb * 32 + kA;
        u.q[0] = *(const uint4*)(p);        // K = kb*32 + kA + 0..7
        u.q[1] = *(const uint4*)(p + 16);   // K = kb*32 + kA + 16..23
        afrag[kb] = u.v;
    }

    // Per-row output scales, hoisted out of the column-tile loop.
    const int rowBase = blockRow + (rowTile << 4) + (hi ? 8 : 0);
    float scale[8];
    #pragma unroll
    for (int g = 0; g < 8; ++g) {
        int row = rowBase + g;
        scale[g] = (row < N) ? rsqrtf(fmaxf(indeg[row], 1.0f)) : 0.0f;
    }
    const bool fullBlock = (blockRow + AROWS) <= N;

    const int kB = hi ? 8 : 0;
    for (int ct = 0; ct < 4; ++ct) {
        const int n = colBase + (ct << 4) + (lane & 15);
        // Buffer all four B fragments, then chain the WMMAs.
        v16bf bfrag[4];
        const unsigned short* bp = Blds + n * LSTRIDE + kB;
        #pragma unroll
        for (int kb = 0; kb < 4; ++kb) {
            BF16x16 u;
            u.q[0] = *(const uint4*)(bp + kb * 32);      // K = kb*32 + kB + 0..7
            u.q[1] = *(const uint4*)(bp + kb * 32 + 16); // K = kb*32 + kB + 16..23
            bfrag[kb] = u.v;
        }
        v8f c = {};
        #pragma unroll
        for (int kb = 0; kb < 4; ++kb)
            c = __builtin_amdgcn_wmma_f32_16x16x32_bf16(false, afrag[kb], false, bfrag[kb],
                                                        (short)0, c, false, false);
        // Epilogue: C VGPR g -> row g + (lane>=16 ? 8:0), col = lane&15
        const float bias = br[n];
        float* orow = out + (size_t)rowBase * D + n;
        if (fullBlock) {
            #pragma unroll
            for (int g = 0; g < 8; ++g)
                orow[(size_t)g * D] += scale[g] * c[g] + bias;
        } else {
            #pragma unroll
            for (int g = 0; g < 8; ++g)
                if (rowBase + g < N)
                    orow[(size_t)g * D] += scale[g] * c[g] + bias;
        }
    }
}

extern "C" void kernel_launch(void* const* d_in, const int* in_sizes, int n_in,
                              void* d_out, int out_size, void* d_ws, size_t ws_size,
                              hipStream_t stream) {
    const float* feat = (const float*)d_in[0];   // [N, 128]
    const float* W    = (const float*)d_in[1];   // [R, 128, 128]
    const float* b    = (const float*)d_in[2];   // [R, 128]
    const int*   src  = (const int*)d_in[3];     // [R, E]
    const int*   dst  = (const int*)d_in[4];     // [R, E]

    const int N = in_sizes[0] / D;
    const int R = in_sizes[2] / D;
    const int E = in_sizes[3] / R;

    float* outdeg = (float*)d_ws;                        // [N]
    float* indeg  = outdeg + N;                          // [N]
    float* agg    = indeg + N;                           // [N, 128]
    unsigned short* Wt = (unsigned short*)(agg + (size_t)N * D); // bf16 [R][n][k]

    (void)hipMemsetAsync(d_out, 0, (size_t)out_size * sizeof(float), stream);

    const int wtot = R * D * D;
    convw_kernel<<<(wtot + 255) / 256, 256, 0, stream>>>(W, Wt, wtot);

    for (int r = 0; r < R; ++r) {
        (void)hipMemsetAsync(outdeg, 0, ((size_t)2 * N + (size_t)N * D) * sizeof(float), stream);
        deg_kernel<<<(E + 255) / 256, 256, 0, stream>>>(
            src + (size_t)r * E, dst + (size_t)r * E, outdeg, indeg, E);
        scatter_kernel<<<(E * 32 + 255) / 256, 256, 0, stream>>>(
            feat, src + (size_t)r * E, dst + (size_t)r * E, outdeg, agg, E);
        gemm_kernel<<<(N + AROWS - 1) / AROWS, 256, 0, stream>>>(
            agg, Wt + (size_t)r * D * D, b + (size_t)r * D, indeg, (float*)d_out, N);
    }
}
